// PointMambaBlock_15676630630609
// MI455X (gfx1250) — matrix operand order repaired and checked by reading, not compiled, not run on hardware
//
#include <hip/hip_runtime.h>
#include <hip/hip_bf16.h>

// ---------------------------------------------------------------------------
// Mamba block on gfx1250 (MI455X): 4x WMMA bf16 GEMMs + lane-parallel scan.
// GEMM wave tile: 32(M) x 64(N) -> 8 v_wmma per K-step, B frags reused 2x.
// ---------------------------------------------------------------------------

#define D_MODEL   512
#define D_STATE   16
#define D_CONV    4
#define D_INNER   1024           // 2 * D_MODEL
#define DT_RANK   32             // (512+15)//16
#define BB        4
#define LL        2048
#define MROWS     (BB * LL)      // 8192

typedef __attribute__((ext_vector_type(16))) __bf16 v16bf;
typedef __attribute__((ext_vector_type(8)))  float  v8f;

union FragU {
    uint4 q[2];
    v16bf v;
};

// A fragment (16x32, bf16, row-major source, lda in elements).
// ISA layout: lanes 0-15 hold row M=lane, K = k0+0..7 (elems 0..7) and
// K = k0+16..23 (elems 8..15); lanes 16-31 hold K = k0+8..15 / k0+24..31.
__device__ __forceinline__ v16bf load_a_frag(const __bf16* A, long lda,
                                             int r0, int k0, int lane) {
    FragU f;
    int r  = r0 + (lane & 15);
    int kb = k0 + ((lane >> 4) << 3);
    const __bf16* p = A + (long)r * lda + kb;
    f.q[0] = *reinterpret_cast<const uint4*>(p);        // K kb..kb+7
    f.q[1] = *reinterpret_cast<const uint4*>(p + 16);   // K kb+16..kb+23
    return f.v;
}

// B fragment (32x16, bf16). Weight stored (N x K) row-major, ldb in elements.
// Lanes 0-15: column N=c0+lane, K = k0+0..7 / k0+16..23; lanes 16-31: K = k0+8..15 / k0+24..31.
__device__ __forceinline__ v16bf load_b_frag(const __bf16* Bw, long ldb,
                                             int c0, int k0, int lane) {
    FragU f;
    int c  = c0 + (lane & 15);
    int kb = k0 + ((lane >> 4) << 3);
    const __bf16* p = Bw + (long)c * ldb + kb;
    f.q[0] = *reinterpret_cast<const uint4*>(p);
    f.q[1] = *reinterpret_cast<const uint4*>(p + 16);
    return f.v;
}

__device__ __forceinline__ float softplus_f(float x) {
    return (x > 20.f) ? x : log1pf(__expf(x));
}
__device__ __forceinline__ float silu_f(float x) {
    return x / (1.f + __expf(-x));
}

// ---------------------------------------------------------------------------
// Generic WMMA GEMM: C[m,n] = sum_k A[m,k] * Bw[n,k]   (A: MxK, Bw: NxK)
// One wave computes a 32(M) x 64(N) tile; 8 v_wmma per K-step.
// mode 0: plain f32 store.   mode 1: C = softplus(acc + bias[n]).
// ---------------------------------------------------------------------------
__global__ __launch_bounds__(128) void k_wmma_gemm(
    const __bf16* __restrict__ A, const __bf16* __restrict__ Bw,
    float* __restrict__ C, int M, int N, int K,
    int lda, int ldb, int ldc, int mode, const float* __restrict__ bias)
{
    int lane = threadIdx.x & 31;
    int wave = threadIdx.x >> 5;
    int tiles_n = N >> 6;
    int tile = blockIdx.x * 4 + wave;
    int total = (M >> 5) * tiles_n;
    if (tile >= total) return;
    int m0 = (tile / tiles_n) << 5;
    int n0 = (tile % tiles_n) << 6;

    v8f acc00 = {}, acc01 = {}, acc02 = {}, acc03 = {};
    v8f acc10 = {}, acc11 = {}, acc12 = {}, acc13 = {};
    for (int k0 = 0; k0 < K; k0 += 32) {
        v16bf a0 = load_a_frag(A, lda, m0,      k0, lane);
        v16bf a1 = load_a_frag(A, lda, m0 + 16, k0, lane);
        v16bf b0 = load_b_frag(Bw, ldb, n0 +  0, k0, lane);
        v16bf b1 = load_b_frag(Bw, ldb, n0 + 16, k0, lane);
        v16bf b2 = load_b_frag(Bw, ldb, n0 + 32, k0, lane);
        v16bf b3 = load_b_frag(Bw, ldb, n0 + 48, k0, lane);
        acc00 = __builtin_amdgcn_wmma_f32_16x16x32_bf16(false, a0, false, b0,
                                                        (short)0, acc00, false, false);
        acc01 = __builtin_amdgcn_wmma_f32_16x16x32_bf16(false, a0, false, b1,
                                                        (short)0, acc01, false, false);
        acc02 = __builtin_amdgcn_wmma_f32_16x16x32_bf16(false, a0, false, b2,
                                                        (short)0, acc02, false, false);
        acc03 = __builtin_amdgcn_wmma_f32_16x16x32_bf16(false, a0, false, b3,
                                                        (short)0, acc03, false, false);
        acc10 = __builtin_amdgcn_wmma_f32_16x16x32_bf16(false, a1, false, b0,
                                                        (short)0, acc10, false, false);
        acc11 = __builtin_amdgcn_wmma_f32_16x16x32_bf16(false, a1, false, b1,
                                                        (short)0, acc11, false, false);
        acc12 = __builtin_amdgcn_wmma_f32_16x16x32_bf16(false, a1, false, b2,
                                                        (short)0, acc12, false, false);
        acc13 = __builtin_amdgcn_wmma_f32_16x16x32_bf16(false, a1, false, b3,
                                                        (short)0, acc13, false, false);
    }

    // C/D layout: VGPR v -> row mBase + v + 8*(lane>=16), col n0 + (lane&15).
    int r_off = (lane >> 4) << 3;
    int col   = lane & 15;
    #pragma unroll
    for (int mi = 0; mi < 2; ++mi) {
        #pragma unroll
        for (int v = 0; v < 8; ++v) {
            int r = m0 + mi * 16 + v + r_off;
            float vals[4];
            if (mi == 0) {
                vals[0] = acc00[v]; vals[1] = acc01[v];
                vals[2] = acc02[v]; vals[3] = acc03[v];
            } else {
                vals[0] = acc10[v]; vals[1] = acc11[v];
                vals[2] = acc12[v]; vals[3] = acc13[v];
            }
            #pragma unroll
            for (int t = 0; t < 4; ++t) {
                int n = n0 + t * 16 + col;
                float val = vals[t];
                if (mode == 1) val = softplus_f(val + bias[n]);
                C[(long)r * ldc + n] = val;
            }
        }
    }
}

// ---------------------------------------------------------------------------
// LayerNorm over D_MODEL=512, one wave per row, bf16 output.
// ---------------------------------------------------------------------------
__global__ __launch_bounds__(256) void k_layernorm(
    const float* __restrict__ x, const float* __restrict__ g,
    const float* __restrict__ be, __bf16* __restrict__ xn, int rows)
{
    int wave = threadIdx.x >> 5;
    int lane = threadIdx.x & 31;
    int row  = blockIdx.x * 8 + wave;
    if (row >= rows) return;
    const float* xr = x + (long)row * D_MODEL;
    float v[16];
    float s = 0.f, ss = 0.f;
    #pragma unroll
    for (int i = 0; i < 16; ++i) {
        float t = xr[lane + 32 * i];
        v[i] = t; s += t; ss += t * t;
    }
    #pragma unroll
    for (int m = 16; m >= 1; m >>= 1) {
        s  += __shfl_xor(s,  m, 32);
        ss += __shfl_xor(ss, m, 32);
    }
    float mu  = s * (1.f / D_MODEL);
    float var = ss * (1.f / D_MODEL) - mu * mu;
    float inv = rsqrtf(var + 1e-5f);
    #pragma unroll
    for (int i = 0; i < 16; ++i) {
        int c = lane + 32 * i;
        float t = (v[i] - mu) * inv * g[c] + be[c];
        xn[(long)row * D_MODEL + c] = (__bf16)t;
    }
}

// ---------------------------------------------------------------------------
// Depthwise causal conv (D_CONV=4) + bias + SiLU over xi = xz[..., :1024].
// ---------------------------------------------------------------------------
__global__ __launch_bounds__(256) void k_conv(
    const float* __restrict__ xz, const float* __restrict__ conv_w,
    const float* __restrict__ conv_b, float* __restrict__ xc,
    __bf16* __restrict__ xcb)
{
    long idx = (long)blockIdx.x * 256 + threadIdx.x;   // over MROWS * D_INNER
    int  d   = (int)(idx & (D_INNER - 1));
    long row = idx >> 10;
    int  l   = (int)(row & (LL - 1));
    const float* w = conv_w + (long)d * D_CONV;        // (D_INNER,1,4)
    float acc = conv_b[d];
    #pragma unroll
    for (int j = 0; j < D_CONV; ++j) {
        int lj = l - (D_CONV - 1) + j;
        if (lj >= 0)
            acc += w[j] * xz[(row + (long)(lj - l)) * (2 * D_INNER) + d];
    }
    float out = silu_f(acc);
    xc[idx]  = out;
    xcb[idx] = (__bf16)out;
}

// ---------------------------------------------------------------------------
// f32 -> bf16 conversions.
// ---------------------------------------------------------------------------
__global__ __launch_bounds__(256) void k_cvt(
    const float* __restrict__ src, __bf16* __restrict__ dst, long n)
{
    long i = (long)blockIdx.x * 256 + threadIdx.x;
    if (i < n) dst[i] = (__bf16)src[i];
}

__global__ __launch_bounds__(256) void k_cvt_slice(
    const float* __restrict__ src, __bf16* __restrict__ dst,
    int rows, int cols, int sld, int dld)
{
    long i = (long)blockIdx.x * 256 + threadIdx.x;
    if (i >= (long)rows * cols) return;
    int r = (int)(i / cols);
    int c = (int)(i % cols);
    dst[(long)r * dld + c] = (__bf16)src[(long)r * sld + c];
}

// ---------------------------------------------------------------------------
// Selective scan: one lane per (b, d, s); y-reduction over the 16 states via
// a shfl_xor butterfly inside each 16-lane group. Lane s==0 applies the
// D skip connection + SiLU(z) gate and stores bf16 for the output GEMM.
// ---------------------------------------------------------------------------
__global__ __launch_bounds__(256) void k_scan(
    const float* __restrict__ dt, const float* __restrict__ u,
    const float* __restrict__ xdbl, const float* __restrict__ xz,
    const float* __restrict__ A_log, const float* __restrict__ Dw,
    __bf16* __restrict__ yg)
{
    int t = blockIdx.x * 256 + threadIdx.x;    // 65536 threads total
    int s = t & 15;
    int d = (t >> 4) & (D_INNER - 1);
    int b = t >> 14;

    float Ads = -__expf(A_log[d * D_STATE + s]);
    float Dd  = Dw[d];
    float h   = 0.f;
    long base = (long)b * LL;

    for (int l = 0; l < LL; ++l) {
        long rb  = base + l;
        float dtv = dt[rb * D_INNER + d];
        float uv  = u[rb * D_INNER + d];
        float Bv  = xdbl[rb * 64 + DT_RANK + s];
        float Cv  = xdbl[rb * 64 + DT_RANK + D_STATE + s];
        h = __expf(dtv * Ads) * h + dtv * Bv * uv;
        float p = h * Cv;
        p += __shfl_xor(p, 1, 32);
        p += __shfl_xor(p, 2, 32);
        p += __shfl_xor(p, 4, 32);
        p += __shfl_xor(p, 8, 32);
        if (s == 0) {
            float yb = p + uv * Dd;
            float zv = xz[rb * (2 * D_INNER) + D_INNER + d];
            yg[rb * D_INNER + d] = (__bf16)(yb * silu_f(zv));
        }
    }
}

// ---------------------------------------------------------------------------
// Host launcher.
// ---------------------------------------------------------------------------
extern "C" void kernel_launch(void* const* d_in, const int* in_sizes, int n_in,
                              void* d_out, int out_size, void* d_ws, size_t ws_size,
                              hipStream_t stream) {
    const float* x      = (const float*)d_in[0];
    const float* ln_g   = (const float*)d_in[1];
    const float* ln_b   = (const float*)d_in[2];
    const float* W_in   = (const float*)d_in[3];
    const float* conv_w = (const float*)d_in[4];
    const float* conv_b = (const float*)d_in[5];
    const float* W_x    = (const float*)d_in[6];
    const float* W_dt   = (const float*)d_in[7];
    const float* b_dt   = (const float*)d_in[8];
    const float* A_log  = (const float*)d_in[9];
    const float* Dw     = (const float*)d_in[10];
    const float* W_out  = (const float*)d_in[11];
    float* out = (float*)d_out;

    char* ws = (char*)d_ws;
    size_t off = 0;
    auto alloc = [&](size_t bytes) -> void* {
        void* p = ws + off;
        off = (off + bytes + 255) & ~(size_t)255;
        return p;
    };

    __bf16* xn_bf   = (__bf16*)alloc((size_t)MROWS * D_MODEL * 2);
    __bf16* Win_bf  = (__bf16*)alloc((size_t)2 * D_INNER * D_MODEL * 2);
    __bf16* Wx_bf   = (__bf16*)alloc((size_t)64 * D_INNER * 2);
    __bf16* Wdt_bf  = (__bf16*)alloc((size_t)D_INNER * DT_RANK * 2);
    __bf16* Wout_bf = (__bf16*)alloc((size_t)D_MODEL * D_INNER * 2);
    float*  xz      = (float*) alloc((size_t)MROWS * 2 * D_INNER * 4);
    float*  xc_f    = (float*) alloc((size_t)MROWS * D_INNER * 4);
    __bf16* xc_bf   = (__bf16*)alloc((size_t)MROWS * D_INNER * 2);
    float*  xdbl    = (float*) alloc((size_t)MROWS * 64 * 4);
    __bf16* dtin_bf = (__bf16*)alloc((size_t)MROWS * DT_RANK * 2);
    float*  dt_f    = (float*) alloc((size_t)MROWS * D_INNER * 4);
    __bf16* yg_bf   = (__bf16*)alloc((size_t)MROWS * D_INNER * 2);

    // Weight f32 -> bf16.
    k_cvt<<<(2 * D_INNER * D_MODEL) / 256, 256, 0, stream>>>(W_in,  Win_bf,  (long)2 * D_INNER * D_MODEL);
    k_cvt<<<(64 * D_INNER)         / 256, 256, 0, stream>>>(W_x,   Wx_bf,   (long)64 * D_INNER);
    k_cvt<<<(D_INNER * DT_RANK)    / 256, 256, 0, stream>>>(W_dt,  Wdt_bf,  (long)D_INNER * DT_RANK);
    k_cvt<<<(D_MODEL * D_INNER)    / 256, 256, 0, stream>>>(W_out, Wout_bf, (long)D_MODEL * D_INNER);

    // LayerNorm -> xn (bf16).
    k_layernorm<<<MROWS / 8, 256, 0, stream>>>(x, ln_g, ln_b, xn_bf, MROWS);

    // GEMM1: xz = xn @ W_in^T  (8192 x 2048, K=512)
    k_wmma_gemm<<<(MROWS / 32) * (2 * D_INNER / 64) / 4, 128, 0, stream>>>(
        xn_bf, Win_bf, xz, MROWS, 2 * D_INNER, D_MODEL,
        D_MODEL, D_MODEL, 2 * D_INNER, 0, nullptr);

    // Causal depthwise conv + SiLU -> xc (f32 + bf16).
    k_conv<<<(MROWS * D_INNER) / 256, 256, 0, stream>>>(xz, conv_w, conv_b, xc_f, xc_bf);

    // GEMM2: x_dbl = xc @ W_x^T  (8192 x 64, K=1024)
    k_wmma_gemm<<<(MROWS / 32) * (64 / 64) / 4, 128, 0, stream>>>(
        xc_bf, Wx_bf, xdbl, MROWS, 64, D_INNER,
        D_INNER, D_INNER, 64, 0, nullptr);

    // bf16 copy of x_dbl[:, :DT_RANK] for the dt projection.
    k_cvt_slice<<<(MROWS * DT_RANK) / 256, 256, 0, stream>>>(
        xdbl, dtin_bf, MROWS, DT_RANK, 64, DT_RANK);

    // GEMM3: dt = softplus(x_dbl[:,:32] @ W_dt^T + b_dt)  (8192 x 1024, K=32)
    k_wmma_gemm<<<(MROWS / 32) * (D_INNER / 64) / 4, 128, 0, stream>>>(
        dtin_bf, Wdt_bf, dt_f, MROWS, D_INNER, DT_RANK,
        DT_RANK, DT_RANK, D_INNER, 1, b_dt);

    // Selective scan + D skip + SiLU(z) gate -> yg (bf16).
    k_scan<<<(BB * D_INNER * D_STATE) / 256, 256, 0, stream>>>(
        dt_f, xc_f, xdbl, xz, A_log, Dw, yg_bf);

    // GEMM4: out = yg @ W_out^T  (8192 x 512, K=1024)
    k_wmma_gemm<<<(MROWS / 32) * (D_MODEL / 64) / 4, 128, 0, stream>>>(
        yg_bf, Wout_bf, out, MROWS, D_MODEL, D_INNER,
        D_INNER, D_INNER, D_MODEL, 0, nullptr);
}
